// CausalSelfAttention_34419867910887
// MI455X (gfx1250) — matrix operand order repaired and checked by reading, not compiled, not run on hardware
//
#include <hip/hip_runtime.h>
#include <hip/hip_bf16.h>
#include <stdint.h>

// ---------------------------------------------------------------------------
// Types for CDNA5 WMMA (gfx1250, wave32)
// ---------------------------------------------------------------------------
typedef __attribute__((ext_vector_type(16))) __bf16 v16bf;
typedef __attribute__((ext_vector_type(8)))  float  v8f;

union Frag {
    v16bf v;
    uint4 q[2];
};

#define WMMA_BF16(A, B, C) \
    __builtin_amdgcn_wmma_f32_16x16x32_bf16(false, (A), false, (B), (short)0, (C), false, false)

// Problem constants
#define BQ   2
#define TT   2048
#define CC   2048
#define HH   16
#define HD   128
#define DC   512
#define DC1  512
#define DROPE 64
#define DNOPE 64
#define MM   4
#define EPSR 1.1920929e-07f
#define BT   (BQ * TT)           // 4096
#define NDOWN (DC + DC1 + DROPE) // 1088
#define NKV  (HH * (DNOPE + HD)) // 3072
#define NGATE (HH * (MM + 1))    // 80

// Async global->LDS copy of 16 bytes (CDNA5 GLOBAL_LOAD_ASYNC_TO_LDS_B128).
// ldsaddr: 32-bit LDS byte address (low 32 bits of generic pointer);
// gaddr:   64-bit global address. Tracked by ASYNCcnt.
__device__ __forceinline__ void async_copy_b128(uint32_t ldsaddr, uint64_t gaddr) {
    asm volatile("global_load_async_to_lds_b128 %0, %1, off"
                 :: "v"(ldsaddr), "v"(gaddr) : "memory");
}
__device__ __forceinline__ void async_wait0() {
    asm volatile("s_wait_asynccnt 0x0" ::: "memory");
}

// ---------------------------------------------------------------------------
// fp32 -> bf16 conversion
// ---------------------------------------------------------------------------
__global__ __launch_bounds__(256) void k_cvt_bf16(const float* __restrict__ src,
                                                  __bf16* __restrict__ dst, int n) {
    int i = (blockIdx.x * 256 + threadIdx.x) * 4;
    if (i + 3 < n) {
        float4 f = *(const float4*)(src + i);
        dst[i + 0] = (__bf16)f.x;
        dst[i + 1] = (__bf16)f.y;
        dst[i + 2] = (__bf16)f.z;
        dst[i + 3] = (__bf16)f.w;
    } else {
        for (int j = i; j < n; ++j) dst[j] = (__bf16)src[j];
    }
}

// ---------------------------------------------------------------------------
// Generic bf16 WMMA GEMM:  C[M,N] (f32) = A[M,K] (bf16) x B[K,N] (bf16)
// Block tile 128x128x32, 256 threads = 8 waves (4x2), wave tile 32x64.
// Double-buffered software pipeline: A tile k+1 staged via async global->LDS
// b128 copies and B tile k+1 loaded to registers while tile k is computed;
// one s_wait_asynccnt + barrier per K-step. N bounds handled by a uniform
// per-block fast/slow path (all N used here are multiples of 8).
// ---------------------------------------------------------------------------
#define GBM 128
#define GBN 128
#define GBK 32

__global__ __launch_bounds__(256) void k_gemm_bf16(
    const __bf16* __restrict__ A, const __bf16* __restrict__ B,
    float* __restrict__ C, int M, int N, int K, int lda, int ldb, int ldc) {
    __shared__ __attribute__((aligned(16))) __bf16 As[2][GBM][GBK + 8];
    __shared__ __attribute__((aligned(16))) __bf16 Bs[2][GBN][GBK + 8];

    const int tid  = threadIdx.x;
    const int lane = tid & 31;
    const int wave = tid >> 5;
    const int half = lane >> 4;
    const int l15  = lane & 15;
    const int wm   = wave >> 1;   // 0..3
    const int wn   = wave & 1;    // 0..1
    const int mblock = blockIdx.y * GBM;
    const int nblock = blockIdx.x * GBN;
    const bool nfull = (nblock + GBN) <= N;

    v8f acc[2][4];
#pragma unroll
    for (int mt = 0; mt < 2; ++mt)
#pragma unroll
        for (int nt = 0; nt < 4; ++nt)
#pragma unroll
            for (int g = 0; g < 8; ++g) acc[mt][nt][g] = 0.0f;

    // Per-thread staging coordinates.
    // A tile: 128x32 bf16 = 512 16B chunks, 2 per thread.
    const int arow0 = tid >> 2;            // chunk tid      : rows 0..63
    const int aks0  = (tid & 3) * 8;
    const int arow1 = (tid + 256) >> 2;    // chunk tid+256  : rows 64..127
    const int aks1  = aks0;
    // B tile: 32x128 as 512 8-wide chunks, 2 per thread.
    const int bkk0 = tid >> 4;             // 0..15
    const int bn0  = (tid & 15) * 8;
    const int bkk1 = (tid + 256) >> 4;     // 16..31
    const int bn1  = bn0;

    auto stage_a = [&](int k0, int buf) {
        async_copy_b128((uint32_t)(uintptr_t)&As[buf][arow0][aks0],
                        (uint64_t)(uintptr_t)(A + (size_t)(mblock + arow0) * lda + k0 + aks0));
        async_copy_b128((uint32_t)(uintptr_t)&As[buf][arow1][aks1],
                        (uint64_t)(uintptr_t)(A + (size_t)(mblock + arow1) * lda + k0 + aks1));
    };
    auto load_b = [&](int k0, uint4* u) {
        if (nfull) {
            u[0] = *(const uint4*)(B + (size_t)(k0 + bkk0) * ldb + nblock + bn0);
            u[1] = *(const uint4*)(B + (size_t)(k0 + bkk1) * ldb + nblock + bn1);
            if (k0 + GBK < K)
                __builtin_prefetch(B + (size_t)(k0 + GBK + bkk0) * ldb + nblock + bn0, 0, 1);
        } else {
            u[0] = make_uint4(0u, 0u, 0u, 0u);
            u[1] = make_uint4(0u, 0u, 0u, 0u);
            if (nblock + bn0 + 7 < N)
                u[0] = *(const uint4*)(B + (size_t)(k0 + bkk0) * ldb + nblock + bn0);
            if (nblock + bn1 + 7 < N)
                u[1] = *(const uint4*)(B + (size_t)(k0 + bkk1) * ldb + nblock + bn1);
        }
    };
    auto store_b = [&](const uint4* u, int buf) {
        const __bf16* e0 = (const __bf16*)&u[0];
        const __bf16* e1 = (const __bf16*)&u[1];
#pragma unroll
        for (int i = 0; i < 8; ++i) Bs[buf][bn0 + i][bkk0] = e0[i];
#pragma unroll
        for (int i = 0; i < 8; ++i) Bs[buf][bn1 + i][bkk1] = e1[i];
    };

    // ---- prologue: stage tile 0 into buffer 0
    uint4 breg[2];
    stage_a(0, 0);
    load_b(0, breg);
    store_b(breg, 0);
    async_wait0();
    __syncthreads();

    const int nsteps = K / GBK;
    for (int step = 0; step < nsteps; ++step) {
        const int buf  = step & 1;
        const bool more = (step + 1) < nsteps;

        // ---- kick off next tile while this one computes
        if (more) {
            stage_a((step + 1) * GBK, buf ^ 1);
            load_b((step + 1) * GBK, breg);
        }

        // ---- fragments (documented CDNA5 16-bit A/B layouts) + WMMA
        Frag fa[2];
#pragma unroll
        for (int mt = 0; mt < 2; ++mt) {
            int row = wm * 32 + mt * 16 + l15;
            int kb  = half * 8;
            fa[mt].q[0] = *(const uint4*)&As[buf][row][kb];
            fa[mt].q[1] = *(const uint4*)&As[buf][row][kb + 16];
        }
        Frag fb[4];
#pragma unroll
        for (int nt = 0; nt < 4; ++nt) {
            int col = wn * 64 + nt * 16 + l15;
            int kb  = half * 8;
            fb[nt].q[0] = *(const uint4*)&Bs[buf][col][kb];
            fb[nt].q[1] = *(const uint4*)&Bs[buf][col][kb + 16];
        }
#pragma unroll
        for (int mt = 0; mt < 2; ++mt)
#pragma unroll
            for (int nt = 0; nt < 4; ++nt)
                acc[mt][nt] = WMMA_BF16(fa[mt].v, fb[nt].v, acc[mt][nt]);

        // ---- finish next tile staging, one barrier per step
        if (more) store_b(breg, buf ^ 1);
        async_wait0();
        __syncthreads();
    }

    // ---- store C (f32), col-guarded
#pragma unroll
    for (int mt = 0; mt < 2; ++mt)
#pragma unroll
        for (int nt = 0; nt < 4; ++nt)
#pragma unroll
            for (int g = 0; g < 8; ++g) {
                int row = mblock + wm * 32 + mt * 16 + g + half * 8;
                int col = nblock + wn * 64 + nt * 16 + l15;
                if (col < N) C[(size_t)row * ldc + col] = acc[mt][nt][g];
            }
}

// ---------------------------------------------------------------------------
// Fused rope + gate-mix + rmsnorm -> bf16 Q/K/V in [B,H,T,HD] layout
// One block per (b,t) token, 128 threads (one per head-dim element).
// ---------------------------------------------------------------------------
__device__ inline float block_sum128(float v, float* red) {
#pragma unroll
    for (int m = 16; m >= 1; m >>= 1) v += __shfl_xor(v, m, 32);
    int wid = threadIdx.x >> 5;
    if ((threadIdx.x & 31) == 0) red[wid] = v;
    __syncthreads();
    v = red[0] + red[1] + red[2] + red[3];
    __syncthreads();
    return v;
}

__global__ __launch_bounds__(128) void k_build_qkv(
    const float* __restrict__ down, const float* __restrict__ qraw,
    const float* __restrict__ kvb, const float* __restrict__ gates_raw,
    const float* __restrict__ ve, const float* __restrict__ cosb,
    const float* __restrict__ sinb, __bf16* __restrict__ Q,
    __bf16* __restrict__ Kb, __bf16* __restrict__ V) {
    __shared__ float red[4];
    __shared__ float krope[DROPE];
    __shared__ float gsh[NGATE];

    const int row = blockIdx.x;          // b*T + t
    const int b   = row / TT;
    const int t   = row - b * TT;
    const int d   = threadIdx.x;

    // k_rope = rope(down[:, 1024:1088])
    if (d < DROPE / 2) {
        float x1 = down[(size_t)row * NDOWN + DC + DC1 + d];
        float x2 = down[(size_t)row * NDOWN + DC + DC1 + DROPE / 2 + d];
        float c  = cosb[(size_t)row * (DROPE / 2) + d];
        float s  = sinb[(size_t)row * (DROPE / 2) + d];
        krope[d]             =  x1 * c + x2 * s;
        krope[DROPE / 2 + d] = -x1 * s + x2 * c;
    }
    if (d < NGATE)
        gsh[d] = 2.0f / (1.0f + __expf(-gates_raw[(size_t)row * NGATE + d]));
    __syncthreads();

    for (int h = 0; h < HH; ++h) {
        const size_t qkv_base = (((size_t)b * HH + h) * TT + t) * HD;

        // ---- K = rmsnorm(concat(k_nope, k_rope))
        float kd = (d < DNOPE) ? kvb[(size_t)row * NKV + h * (DNOPE + HD) + d]
                               : krope[d - DNOPE];
        float ksum = block_sum128(kd * kd, red);
        Kb[qkv_base + d] = (__bf16)(kd * rsqrtf(ksum * (1.0f / HD) + EPSR));

        // ---- V = g0 * v + sum_m g[1+m] * ve[m]
        float vv = gsh[h * (MM + 1)] *
                   kvb[(size_t)row * NKV + h * (DNOPE + HD) + DNOPE + d];
#pragma unroll
        for (int m = 0; m < MM; ++m)
            vv += gsh[h * (MM + 1) + 1 + m] *
                  ve[((size_t)row * MM + m) * (HH * HD) + h * HD + d];
        V[qkv_base + d] = (__bf16)vv;

        // ---- Q = rmsnorm(concat(q_nope, rope(q_rope)))
        const size_t qb = (size_t)row * (HH * HD) + h * HD;
        float qv;
        if (d < DNOPE) {
            qv = qraw[qb + d];
        } else {
            int i = d - DNOPE;
            if (i < DROPE / 2) {
                float c = cosb[(size_t)row * (DROPE / 2) + i];
                float s = sinb[(size_t)row * (DROPE / 2) + i];
                qv = qraw[qb + DNOPE + i] * c + qraw[qb + DNOPE + DROPE / 2 + i] * s;
            } else {
                int j = i - DROPE / 2;
                float c = cosb[(size_t)row * (DROPE / 2) + j];
                float s = sinb[(size_t)row * (DROPE / 2) + j];
                qv = -qraw[qb + DNOPE + j] * s + qraw[qb + DNOPE + DROPE / 2 + j] * c;
            }
        }
        float qsum = block_sum128(qv * qv, red);
        Q[qkv_base + d] = (__bf16)(qv * rsqrtf(qsum * (1.0f / HD) + EPSR));
    }
}

// ---------------------------------------------------------------------------
// Flash attention (causal). Block = (q-tile of 128, h, b); 8 waves x 16 rows.
// Q.K^T via WMMA (K B-fragments straight from global, row-major K is already
// K-contiguous per column). Online softmax in C-layout with shfl_xor row
// reductions. P staged to LDS (bf16) and re-loaded in A-layout; V tile staged
// transposed in LDS for B-layout reads. Output written directly as bf16 Y.
// ---------------------------------------------------------------------------
__global__ __launch_bounds__(256) void k_attn(
    const __bf16* __restrict__ Q, const __bf16* __restrict__ Kb,
    const __bf16* __restrict__ V, __bf16* __restrict__ Y) {
    __shared__ __attribute__((aligned(16))) __bf16 Vt[HD][64 + 8];
    __shared__ __attribute__((aligned(16))) __bf16 Pl[8][16][64 + 8];

    const int b  = blockIdx.z;
    const int h  = blockIdx.y;
    const int q0 = blockIdx.x * 128;
    const int tid  = threadIdx.x;
    const int lane = tid & 31;
    const int w    = tid >> 5;
    const int half = lane >> 4;
    const int l15  = lane & 15;
    const size_t bh = ((size_t)b * HH + h) * TT;

    // Q A-fragments: 16 rows (this wave) x 128 HD, 4 K-chunks of 32
    Frag qa[4];
    {
        const __bf16* qrow = Q + (bh + q0 + w * 16 + l15) * HD;
#pragma unroll
        for (int c = 0; c < 4; ++c) {
            int kb = c * 32 + half * 8;
            qa[c].q[0] = *(const uint4*)(qrow + kb);
            qa[c].q[1] = *(const uint4*)(qrow + kb + 16);
        }
    }

    v8f o[8];
#pragma unroll
    for (int nt = 0; nt < 8; ++nt)
#pragma unroll
        for (int g = 0; g < 8; ++g) o[nt][g] = 0.0f;
    float mrow[8], lrow[8];
#pragma unroll
    for (int g = 0; g < 8; ++g) { mrow[g] = -1e30f; lrow[g] = 0.0f; }

    const float scale = 0.08838834764831845f;   // 1/sqrt(128)
    const int irow0 = q0 + w * 16 + half * 8;
    const int ntiles = (q0 + 128) / 64;

    for (int jt = 0; jt < ntiles; ++jt) {
        const int j0 = jt * 64;
        __syncthreads();
        // ---- stage V tile transposed: Vt[d][key]
#pragma unroll
        for (int it = 0; it < 4; ++it) {
            int idx = tid + it * 256;        // 0..1023
            int kk  = idx >> 4;              // key 0..63
            int d0  = (idx & 15) * 8;        // d start
            uint4 u = *(const uint4*)(V + (bh + j0 + kk) * HD + d0);
            const __bf16* e = (const __bf16*)&u;
#pragma unroll
            for (int i = 0; i < 8; ++i) Vt[d0 + i][kk] = e[i];
        }
        __syncthreads();

        // ---- S = Q . K^T   (16 x 64 per wave, 16 WMMA)
        v8f s[4];
#pragma unroll
        for (int nt = 0; nt < 4; ++nt)
#pragma unroll
            for (int g = 0; g < 8; ++g) s[nt][g] = 0.0f;
#pragma unroll
        for (int c = 0; c < 4; ++c) {
#pragma unroll
            for (int nt = 0; nt < 4; ++nt) {
                Frag kf;
                const __bf16* krow = Kb + (bh + j0 + nt * 16 + l15) * HD + c * 32 + half * 8;
                kf.q[0] = *(const uint4*)(krow);
                kf.q[1] = *(const uint4*)(krow + 16);
                s[nt] = WMMA_BF16(qa[c].v, kf.v, s[nt]);
            }
        }

        // ---- scale + causal mask + online softmax (C-layout)
#pragma unroll
        for (int g = 0; g < 8; ++g) {
            const int irow = irow0 + g;
            float mx = -1e30f;
#pragma unroll
            for (int nt = 0; nt < 4; ++nt) {
                int j = j0 + nt * 16 + l15;
                float val = s[nt][g] * scale;
                if (j > irow) val = -1e30f;
                s[nt][g] = val;
                mx = fmaxf(mx, val);
            }
#pragma unroll
            for (int mk = 8; mk >= 1; mk >>= 1) mx = fmaxf(mx, __shfl_xor(mx, mk, 32));
            float mnew  = fmaxf(mrow[g], mx);
            float alpha = __expf(mrow[g] - mnew);
            float rs = 0.0f;
#pragma unroll
            for (int nt = 0; nt < 4; ++nt) {
                float p = __expf(s[nt][g] - mnew);
                s[nt][g] = p;
                rs += p;
            }
#pragma unroll
            for (int mk = 8; mk >= 1; mk >>= 1) rs += __shfl_xor(rs, mk, 32);
            lrow[g] = lrow[g] * alpha + rs;
            mrow[g] = mnew;
#pragma unroll
            for (int nt = 0; nt < 8; ++nt) o[nt][g] *= alpha;
        }

        // ---- P -> LDS (wave-private), re-load in A-layout
#pragma unroll
        for (int nt = 0; nt < 4; ++nt)
#pragma unroll
            for (int g = 0; g < 8; ++g)
                Pl[w][g + half * 8][nt * 16 + l15] = (__bf16)s[nt][g];

#pragma unroll
        for (int c = 0; c < 2; ++c) {
            Frag pa;
            int kb0 = c * 32 + half * 8;
            pa.q[0] = *(const uint4*)&Pl[w][l15][kb0];
            pa.q[1] = *(const uint4*)&Pl[w][l15][kb0 + 16];
#pragma unroll
            for (int nt = 0; nt < 8; ++nt) {
                Frag vf;
                int dcol = nt * 16 + l15;
                int kk   = c * 32 + half * 8;
                vf.q[0] = *(const uint4*)&Vt[dcol][kk];
                vf.q[1] = *(const uint4*)&Vt[dcol][kk + 16];
                o[nt] = WMMA_BF16(pa.v, vf.v, o[nt]);
            }
        }
    }

    // ---- epilogue: normalize, store Y as bf16 [B*T][H*HD]
#pragma unroll
    for (int nt = 0; nt < 8; ++nt)
#pragma unroll
        for (int g = 0; g < 8; ++g) {
            int i = irow0 + g;
            float val = o[nt][g] / lrow[g];
            Y[((size_t)b * TT + i) * (HH * HD) + h * HD + nt * 16 + l15] = (__bf16)val;
        }
}

// ---------------------------------------------------------------------------
// Host launcher
// ---------------------------------------------------------------------------
static inline void* ws_alloc(char* base, size_t& off, size_t bytes) {
    void* p = base + off;
    off += bytes;
    off = (off + 255) & ~(size_t)255;
    return p;
}

extern "C" void kernel_launch(void* const* d_in, const int* in_sizes, int n_in,
                              void* d_out, int out_size, void* d_ws, size_t ws_size,
                              hipStream_t stream) {
    const float* x      = (const float*)d_in[0];
    const float* ve     = (const float*)d_in[1];
    const float* cosb   = (const float*)d_in[2];
    const float* sinb   = (const float*)d_in[3];
    const float* w_down = (const float*)d_in[4];
    const float* w_ukv  = (const float*)d_in[5];
    const float* w_uq   = (const float*)d_in[6];
    const float* w_gate = (const float*)d_in[7];
    const float* w_proj = (const float*)d_in[8];
    (void)in_sizes; (void)n_in; (void)out_size; (void)ws_size;

    char* ws = (char*)d_ws;
    size_t off = 0;
    __bf16* x_bf     = (__bf16*)ws_alloc(ws, off, (size_t)BT * CC * 2);
    __bf16* wdown_bf = (__bf16*)ws_alloc(ws, off, (size_t)CC * NDOWN * 2);
    __bf16* wukv_bf  = (__bf16*)ws_alloc(ws, off, (size_t)DC * NKV * 2);
    __bf16* wuq_bf   = (__bf16*)ws_alloc(ws, off, (size_t)DC1 * (HH * HD) * 2);
    __bf16* wgate_bf = (__bf16*)ws_alloc(ws, off, (size_t)CC * NGATE * 2);
    __bf16* wproj_bf = (__bf16*)ws_alloc(ws, off, (size_t)(HH * HD) * CC * 2);
    float*  down_f   = (float*)ws_alloc(ws, off, (size_t)BT * NDOWN * 4);
    __bf16* down_bf  = (__bf16*)ws_alloc(ws, off, (size_t)BT * NDOWN * 2);
    float*  gates_f  = (float*)ws_alloc(ws, off, (size_t)BT * NGATE * 4);
    float*  kv_f     = (float*)ws_alloc(ws, off, (size_t)BT * NKV * 4);
    float*  qraw_f   = (float*)ws_alloc(ws, off, (size_t)BT * (HH * HD) * 4);
    __bf16* Qbf      = (__bf16*)ws_alloc(ws, off, (size_t)BQ * HH * TT * HD * 2);
    __bf16* Kbf      = (__bf16*)ws_alloc(ws, off, (size_t)BQ * HH * TT * HD * 2);
    __bf16* Vbf      = (__bf16*)ws_alloc(ws, off, (size_t)BQ * HH * TT * HD * 2);
    __bf16* Ybf      = (__bf16*)ws_alloc(ws, off, (size_t)BT * (HH * HD) * 2);

    auto cvt = [&](const float* s, __bf16* d, size_t n) {
        int blocks = (int)((n + 1023) / 1024);
        k_cvt_bf16<<<blocks, 256, 0, stream>>>(s, d, (int)n);
    };
    cvt(x,      x_bf,     (size_t)BT * CC);
    cvt(w_down, wdown_bf, (size_t)CC * NDOWN);
    cvt(w_ukv,  wukv_bf,  (size_t)DC * NKV);
    cvt(w_uq,   wuq_bf,   (size_t)DC1 * (HH * HD));
    cvt(w_gate, wgate_bf, (size_t)CC * NGATE);
    cvt(w_proj, wproj_bf, (size_t)(HH * HD) * CC);

    auto gemm = [&](const __bf16* A, const __bf16* B, float* C,
                    int M, int N, int K, int lda, int ldb, int ldc) {
        dim3 grid((N + GBN - 1) / GBN, (M + GBM - 1) / GBM);
        k_gemm_bf16<<<grid, 256, 0, stream>>>(A, B, C, M, N, K, lda, ldb, ldc);
    };

    // down = x @ w_down ; gates_raw = x @ w_gate
    gemm(x_bf, wdown_bf, down_f,  BT, NDOWN, CC, CC, NDOWN, NDOWN);
    gemm(x_bf, wgate_bf, gates_f, BT, NGATE, CC, CC, NGATE, NGATE);

    cvt(down_f, down_bf, (size_t)BT * NDOWN);

    // kv = c_kv @ w_ukv ; qraw = c_q @ w_uq  (column slices of down)
    gemm(down_bf,       wukv_bf, kv_f,   BT, NKV,     DC,  NDOWN, NKV,     NKV);
    gemm(down_bf + DC,  wuq_bf,  qraw_f, BT, HH * HD, DC1, NDOWN, HH * HD, HH * HD);

    // fused rope / gate-mix / rmsnorm -> bf16 Q,K,V [B,H,T,HD]
    k_build_qkv<<<BT, 128, 0, stream>>>(down_f, qraw_f, kv_f, gates_f, ve,
                                        cosb, sinb, Qbf, Kbf, Vbf);

    // flash attention -> Ybf [B*T][H*HD] bf16
    dim3 agrid(TT / 128, HH, BQ);
    k_attn<<<agrid, 256, 0, stream>>>(Qbf, Kbf, Vbf, Ybf);

    // out = Y @ w_proj  (f32)
    gemm(Ybf, wproj_bf, (float*)d_out, BT, CC, HH * HD, HH * HD, CC, CC);
}
